// CustomEncoderLayer_23433341567460
// MI455X (gfx1250) — matrix operand order repaired and checked by reading, not compiled, not run on hardware
//
#include <hip/hip_runtime.h>
#include <hip/hip_bf16.h>

typedef __attribute__((ext_vector_type(16))) __bf16        v16bf;
typedef __attribute__((ext_vector_type(8)))  float         v8f;
typedef __attribute__((ext_vector_type(8)))  unsigned int  v8u;

#define D_MODEL 1024
#define N_HEAD  16
#define D_HEAD  64
#define D_FF    4096
#define MROWS   8192   // B*S = 8*1024

// ---------- helpers ----------
__device__ __forceinline__ unsigned short f2bf(float f) {
    unsigned u = __float_as_uint(f);
    unsigned r = u + 0x7FFFu + ((u >> 16) & 1u);   // round-to-nearest-even
    return (unsigned short)(r >> 16);
}
__device__ __forceinline__ unsigned pack2(float a, float b) {
    return (unsigned)f2bf(a) | ((unsigned)f2bf(b) << 16);
}

// A fragment: 16 bf16 per lane, lane row = lane&15.
// lanes 0-15 hold K = {k0+0..15}; lanes 16-31 = {k0+16..31}
__device__ __forceinline__ v16bf load_a_frag(const unsigned short* arow, int k0, int hi) {
    uint4 c0 = *(const uint4*)(arow + k0 + hi * 16);
    uint4 c1 = *(const uint4*)(arow + k0 + 8 + hi * 16);
    v8u u = { c0.x, c0.y, c0.z, c0.w, c1.x, c1.y, c1.z, c1.w };
    return __builtin_bit_cast(v16bf, u);
}
// B fragment: 16 consecutive bf16 along K (Bt is [N,K] row-major)
__device__ __forceinline__ v16bf load_b_frag(const unsigned short* p) {
    v8u u = *(const v8u*)p;
    return __builtin_bit_cast(v16bf, u);
}
__device__ __forceinline__ v8f wmma_bf16(v16bf a, v16bf b, v8f c) {
    return __builtin_amdgcn_wmma_f32_16x16x32_bf16(false, a, false, b, (short)0, c, false, false);
}

// Wave computes 32(M) x 64(N) f32 accumulators.
// Software pipelined, unrolled x2 with two named fragment sets (no rotation copies).
// A:[M,K] bf16 row-major, Bt:[N,K] bf16 row-major (i.e. W transposed)
__device__ __forceinline__ void gemm_core_2x4(const unsigned short* A, const unsigned short* Bt,
                                              int K, size_t m0, size_t n0, int lane,
                                              v8f acc[2][4]) {
    int r = lane & 15, hi = lane >> 4;
    const unsigned short* arow0 = A + (m0 + (size_t)r) * (size_t)K;
    const unsigned short* arow1 = A + (m0 + 16 + (size_t)r) * (size_t)K;
    const unsigned short* brow  = Bt + (n0 + (size_t)r) * (size_t)K + hi * 16;

    // set C holds step k, set N holds step k+32
    v16bf a0c = load_a_frag(arow0, 0, hi);
    v16bf a1c = load_a_frag(arow1, 0, hi);
    v16bf bc[4];
#pragma unroll
    for (int t = 0; t < 4; ++t) bc[t] = load_b_frag(brow + (size_t)t * 16 * K);

    int k = 0;
    for (; k + 64 < K; k += 64) {
        // load step k+32 into set N
        v16bf a0n = load_a_frag(arow0, k + 32, hi);
        v16bf a1n = load_a_frag(arow1, k + 32, hi);
        v16bf bn[4];
#pragma unroll
        for (int t = 0; t < 4; ++t) bn[t] = load_b_frag(brow + (size_t)t * 16 * K + k + 32);
        __builtin_prefetch(arow0 + k + 128, 0, 3);
        __builtin_prefetch(arow1 + k + 128, 0, 3);
        // compute step k from set C
#pragma unroll
        for (int t = 0; t < 4; ++t) acc[0][t] = wmma_bf16(a0c, bc[t], acc[0][t]);
#pragma unroll
        for (int t = 0; t < 4; ++t) acc[1][t] = wmma_bf16(a1c, bc[t], acc[1][t]);
        // refill set C with step k+64 (direct loads, no copies)
        a0c = load_a_frag(arow0, k + 64, hi);
        a1c = load_a_frag(arow1, k + 64, hi);
#pragma unroll
        for (int t = 0; t < 4; ++t) bc[t] = load_b_frag(brow + (size_t)t * 16 * K + k + 64);
        // compute step k+32 from set N
#pragma unroll
        for (int t = 0; t < 4; ++t) acc[0][t] = wmma_bf16(a0n, bn[t], acc[0][t]);
#pragma unroll
        for (int t = 0; t < 4; ++t) acc[1][t] = wmma_bf16(a1n, bn[t], acc[1][t]);
    }
    // tail: k == K-64; set C holds step k, load and compute k+32
    v16bf a0n = load_a_frag(arow0, k + 32, hi);
    v16bf a1n = load_a_frag(arow1, k + 32, hi);
    v16bf bn[4];
#pragma unroll
    for (int t = 0; t < 4; ++t) bn[t] = load_b_frag(brow + (size_t)t * 16 * K + k + 32);
#pragma unroll
    for (int t = 0; t < 4; ++t) acc[0][t] = wmma_bf16(a0c, bc[t], acc[0][t]);
#pragma unroll
    for (int t = 0; t < 4; ++t) acc[1][t] = wmma_bf16(a1c, bc[t], acc[1][t]);
#pragma unroll
    for (int t = 0; t < 4; ++t) acc[0][t] = wmma_bf16(a0n, bn[t], acc[0][t]);
#pragma unroll
    for (int t = 0; t < 4; ++t) acc[1][t] = wmma_bf16(a1n, bn[t], acc[1][t]);
}

// ---------- weight transpose + f32->bf16 : Wt[n*K+k] = bf(W[k*N+n]) ----------
__global__ __launch_bounds__(256) void transpose_bf16(const float* __restrict__ W,
                                                      unsigned short* __restrict__ Wt,
                                                      int K, int N) {
    __shared__ float tile[16][17];
    int k0 = blockIdx.x * 16, n0 = blockIdx.y * 16;
    int tx = threadIdx.x, ty = threadIdx.y;
    tile[ty][tx] = W[(size_t)(k0 + ty) * N + n0 + tx];
    __syncthreads();
    Wt[(size_t)(n0 + ty) * K + k0 + tx] = f2bf(tile[tx][ty]);
}

// ---------- layernorm: f32 in -> bf16 out ----------
__global__ __launch_bounds__(256) void ln_kernel(const float* __restrict__ x,
                                                 const float* __restrict__ g,
                                                 const float* __restrict__ b,
                                                 unsigned short* __restrict__ out) {
    __shared__ float red[256];
    int row = blockIdx.x, tid = threadIdx.x;
    const float* xr = x + (size_t)row * D_MODEL;
    float s = 0.f;
#pragma unroll
    for (int j = 0; j < 4; ++j) s += xr[tid + j * 256];
    red[tid] = s; __syncthreads();
    for (int off = 128; off > 0; off >>= 1) { if (tid < off) red[tid] += red[tid + off]; __syncthreads(); }
    float mu = red[0] * (1.0f / D_MODEL);
    __syncthreads();
    float v = 0.f;
#pragma unroll
    for (int j = 0; j < 4; ++j) { float d = xr[tid + j * 256] - mu; v += d * d; }
    red[tid] = v; __syncthreads();
    for (int off = 128; off > 0; off >>= 1) { if (tid < off) red[tid] += red[tid + off]; __syncthreads(); }
    float rs = rsqrtf(red[0] * (1.0f / D_MODEL) + 1e-5f);
#pragma unroll
    for (int j = 0; j < 4; ++j) {
        int c = tid + j * 256;
        out[(size_t)row * D_MODEL + c] = f2bf((xr[c] - mu) * rs * g[c] + b[c]);
    }
}

// ---------- QKV projection: X[M,1024] @ W -> Q,K:[B,H,S,64], V^T:[B,H,64,S] ----------
__global__ __launch_bounds__(128) void gemm_qkv(const unsigned short* __restrict__ X,
                                                const unsigned short* WqT, const unsigned short* WkT,
                                                const unsigned short* WvT,
                                                const float* bq, const float* bk, const float* bv,
                                                unsigned short* Qo, unsigned short* Ko,
                                                unsigned short* VTo) {
    int z = blockIdx.z;
    const unsigned short* Wt = (z == 0) ? WqT : (z == 1) ? WkT : WvT;
    const float* bias = (z == 0) ? bq : (z == 1) ? bk : bv;
    int lane = threadIdx.x & 31, wave = threadIdx.x >> 5;
    size_t m0 = ((size_t)blockIdx.y * 4 + wave) * 32;
    size_t n0 = (size_t)blockIdx.x * 64;
    v8f acc[2][4];
#pragma unroll
    for (int rb = 0; rb < 2; ++rb)
#pragma unroll
        for (int t = 0; t < 4; ++t) acc[rb][t] = (v8f){0,0,0,0,0,0,0,0};
    gemm_core_2x4(X, Wt, D_MODEL, m0, n0, lane, acc);
    int r = lane & 15, hi = lane >> 4;
#pragma unroll
    for (int rb = 0; rb < 2; ++rb) {
#pragma unroll
        for (int t = 0; t < 4; ++t) {
#pragma unroll
            for (int e = 0; e < 8; ++e) {
                int n = (int)n0 + t * 16 + r;
                int m = (int)m0 + rb * 16 + e + hi * 8;
                unsigned short w = f2bf(acc[rb][t][e] + bias[n]);
                int bb = m >> 10, s = m & 1023, h = n >> 6, d = n & 63;
                if (z == 0)       Qo [(((size_t)(bb * 16 + h)) * 1024 + s) * 64 + d] = w;
                else if (z == 1)  Ko [(((size_t)(bb * 16 + h)) * 1024 + s) * 64 + d] = w;
                else              VTo[(((size_t)(bb * 16 + h)) * 64 + d) * 1024 + s] = w;
            }
        }
    }
}

// ---------- attention: softmax(QK^T/8) V, one wave = 16 query rows ----------
__global__ __launch_bounds__(32) void attn_kernel(const unsigned short* __restrict__ Q,
                                                  const unsigned short* __restrict__ Km,
                                                  const unsigned short* __restrict__ Vt,
                                                  unsigned short* __restrict__ ctx) {
    __shared__ float sm[16 * 1024];               // 64KB score tile
    int lane = threadIdx.x;
    int bid = blockIdx.x;
    int qt = bid & 63;                            // query tile
    int bh = bid >> 6;                            // b*16+h
    const unsigned short* Qh = Q  + (size_t)bh * 1024 * 64;
    const unsigned short* Kh = Km + (size_t)bh * 1024 * 64;
    const unsigned short* Vh = Vt + (size_t)bh * 64 * 1024;
    int r = lane & 15, hi = lane >> 4;
    int q0 = qt * 16;

    // ---- phase 1: scores = Q Kt / 8 into LDS (pipelined, 2x unrolled) ----
    const unsigned short* arow = Qh + (size_t)(q0 + r) * 64;
    v16bf aq0 = load_a_frag(arow, 0, hi);
    v16bf aq1 = load_a_frag(arow, 32, hi);
    const unsigned short* kbase = Kh + (size_t)r * 64 + hi * 16;

    auto kload = [&](int jb, v16bf& x0, v16bf& x1) {
        const unsigned short* p = kbase + (size_t)jb * 16 * 64;
        x0 = load_b_frag(p);
        x1 = load_b_frag(p + 32);
    };
    auto kcompute = [&](int jb, v16bf x0, v16bf x1) {
        v8f acc = (v8f){0,0,0,0,0,0,0,0};
        acc = wmma_bf16(aq0, x0, acc);
        acc = wmma_bf16(aq1, x1, acc);
        int col = jb * 16 + r;
#pragma unroll
        for (int e = 0; e < 8; ++e) sm[(e + hi * 8) * 1024 + col] = acc[e] * 0.125f;
    };

    v16bf c0, c1, n0, n1;
    kload(0, c0, c1);
    int jb = 0;
    for (; jb + 2 < 64; jb += 2) {
        kload(jb + 1, n0, n1);
        kcompute(jb, c0, c1);
        kload(jb + 2, c0, c1);
        kcompute(jb + 1, n0, n1);
    }
    kload(63, n0, n1);
    kcompute(62, c0, c1);
    kcompute(63, n0, n1);
    __syncthreads();

    // ---- phase 2: softmax over each row (2 lanes per row, 512 cols each) ----
    float* prow = sm + (lane & 15) * 1024 + hi * 512;
    float mx = -1e30f;
    for (int i = 0; i < 512; i += 4) {
        float4 v = *(float4*)(prow + i);
        mx = fmaxf(mx, fmaxf(fmaxf(v.x, v.y), fmaxf(v.z, v.w)));
    }
    mx = fmaxf(mx, __shfl_xor(mx, 16));
    float sum = 0.f;
    for (int i = 0; i < 512; i += 4) {
        float4 v = *(float4*)(prow + i);
        v.x = __expf(v.x - mx); v.y = __expf(v.y - mx);
        v.z = __expf(v.z - mx); v.w = __expf(v.w - mx);
        sum += v.x + v.y + v.z + v.w;
        *(float4*)(prow + i) = v;
    }
    sum += __shfl_xor(sum, 16);
    __syncthreads();

    // ---- phase 3: O = P V (pipelined, 2x unrolled) ----
    v8f o[4];
#pragma unroll
    for (int t = 0; t < 4; ++t) o[t] = (v8f){0,0,0,0,0,0,0,0};
    const unsigned short* vbase = Vh + (size_t)r * 1024 + hi * 16;

    auto vload = [&](int jc, v16bf bvv[4]) {
#pragma unroll
        for (int t = 0; t < 4; ++t)
            bvv[t] = load_b_frag(vbase + (size_t)t * 16 * 1024 + jc * 32);
    };
    auto pcompute = [&](int jc, v16bf bvv[4]) {
        const float* pr = sm + r * 1024 + jc * 32 + hi * 16;
        float4 f0 = *(const float4*)(pr);
        float4 f1 = *(const float4*)(pr + 4);
        float4 f2 = *(const float4*)(pr + 8);
        float4 f3 = *(const float4*)(pr + 12);
        v8u pu = { pack2(f0.x, f0.y), pack2(f0.z, f0.w), pack2(f1.x, f1.y), pack2(f1.z, f1.w),
                   pack2(f2.x, f2.y), pack2(f2.z, f2.w), pack2(f3.x, f3.y), pack2(f3.z, f3.w) };
        v16bf ap = __builtin_bit_cast(v16bf, pu);
#pragma unroll
        for (int t = 0; t < 4; ++t) o[t] = wmma_bf16(ap, bvv[t], o[t]);
    };

    v16bf vc[4], vn[4];
    vload(0, vc);
    int jc = 0;
    for (; jc + 2 < 32; jc += 2) {
        vload(jc + 1, vn);
        pcompute(jc, vc);
        vload(jc + 2, vc);
        pcompute(jc + 1, vn);
    }
    vload(31, vn);
    pcompute(30, vc);
    pcompute(31, vn);

    // ---- epilogue: divide by row sum, scatter to ctx [B,S,1024] ----
    int bb = bh >> 4, h = bh & 15;
#pragma unroll
    for (int t = 0; t < 4; ++t) {
#pragma unroll
        for (int e = 0; e < 8; ++e) {
            int row = e + hi * 8;
            float dn = __shfl(sum, row);
            int q = q0 + row, d = t * 16 + r;
            ctx[((size_t)(bb * 1024 + q)) * 1024 + h * 64 + d] = f2bf(o[t][e] / dn);
        }
    }
}

// ---------- GEMM + bias + residual -> f32 out ----------
__global__ __launch_bounds__(128) void gemm_bias_res_f32(const unsigned short* __restrict__ A,
                                                         const unsigned short* __restrict__ Bt,
                                                         const float* __restrict__ bias,
                                                         const float* __restrict__ res,
                                                         float* __restrict__ out, int K, int N) {
    int lane = threadIdx.x & 31, wave = threadIdx.x >> 5;
    size_t m0 = ((size_t)blockIdx.y * 4 + wave) * 32;
    size_t n0 = (size_t)blockIdx.x * 64;
    v8f acc[2][4];
#pragma unroll
    for (int rb = 0; rb < 2; ++rb)
#pragma unroll
        for (int t = 0; t < 4; ++t) acc[rb][t] = (v8f){0,0,0,0,0,0,0,0};
    gemm_core_2x4(A, Bt, K, m0, n0, lane, acc);
    int r = lane & 15, hi = lane >> 4;
#pragma unroll
    for (int rb = 0; rb < 2; ++rb) {
#pragma unroll
        for (int t = 0; t < 4; ++t) {
#pragma unroll
            for (int e = 0; e < 8; ++e) {
                size_t n = n0 + t * 16 + r;
                size_t m = m0 + rb * 16 + e + hi * 8;
                out[m * N + n] = acc[rb][t][e] + bias[n] + res[m * N + n];
            }
        }
    }
}

// ---------- GEMM + bias + relu -> bf16 out ----------
__global__ __launch_bounds__(128) void gemm_bias_relu_bf16(const unsigned short* __restrict__ A,
                                                           const unsigned short* __restrict__ Bt,
                                                           const float* __restrict__ bias,
                                                           unsigned short* __restrict__ out,
                                                           int K, int N) {
    int lane = threadIdx.x & 31, wave = threadIdx.x >> 5;
    size_t m0 = ((size_t)blockIdx.y * 4 + wave) * 32;
    size_t n0 = (size_t)blockIdx.x * 64;
    v8f acc[2][4];
#pragma unroll
    for (int rb = 0; rb < 2; ++rb)
#pragma unroll
        for (int t = 0; t < 4; ++t) acc[rb][t] = (v8f){0,0,0,0,0,0,0,0};
    gemm_core_2x4(A, Bt, K, m0, n0, lane, acc);
    int r = lane & 15, hi = lane >> 4;
#pragma unroll
    for (int rb = 0; rb < 2; ++rb) {
#pragma unroll
        for (int t = 0; t < 4; ++t) {
#pragma unroll
            for (int e = 0; e < 8; ++e) {
                size_t n = n0 + t * 16 + r;
                size_t m = m0 + rb * 16 + e + hi * 8;
                out[m * N + n] = f2bf(fmaxf(acc[rb][t][e] + bias[n], 0.f));
            }
        }
    }
}

// ---------- host launch ----------
extern "C" void kernel_launch(void* const* d_in, const int* in_sizes, int n_in,
                              void* d_out, int out_size, void* d_ws, size_t ws_size,
                              hipStream_t stream) {
    (void)in_sizes; (void)n_in; (void)out_size; (void)ws_size;
    const float* src   = (const float*)d_in[0];
    const float* Wq    = (const float*)d_in[1];
    const float* bq    = (const float*)d_in[2];
    const float* Wk    = (const float*)d_in[3];
    const float* bk    = (const float*)d_in[4];
    const float* Wv    = (const float*)d_in[5];
    const float* bv    = (const float*)d_in[6];
    const float* Wo    = (const float*)d_in[7];
    const float* bo    = (const float*)d_in[8];
    const float* W1    = (const float*)d_in[9];
    const float* b1    = (const float*)d_in[10];
    const float* W2    = (const float*)d_in[11];
    const float* b2    = (const float*)d_in[12];
    const float* g1    = (const float*)d_in[13];
    const float* beta1 = (const float*)d_in[14];
    const float* g2    = (const float*)d_in[15];
    const float* beta2 = (const float*)d_in[16];
    float* out = (float*)d_out;

    char* w = (char*)d_ws;
    size_t off = 0;
    auto alloc = [&](size_t bytes) { void* p = w + off; off = (off + bytes + 255) & ~(size_t)255; return p; };
    unsigned short* xh   = (unsigned short*)alloc((size_t)MROWS * D_MODEL * 2);
    unsigned short* WqT  = (unsigned short*)alloc((size_t)D_MODEL * D_MODEL * 2);
    unsigned short* WkT  = (unsigned short*)alloc((size_t)D_MODEL * D_MODEL * 2);
    unsigned short* WvT  = (unsigned short*)alloc((size_t)D_MODEL * D_MODEL * 2);
    unsigned short* WoT  = (unsigned short*)alloc((size_t)D_MODEL * D_MODEL * 2);
    unsigned short* W1T  = (unsigned short*)alloc((size_t)D_MODEL * D_FF * 2);
    unsigned short* W2T  = (unsigned short*)alloc((size_t)D_FF * D_MODEL * 2);
    unsigned short* qb   = (unsigned short*)alloc((size_t)MROWS * D_MODEL * 2);
    unsigned short* kb   = (unsigned short*)alloc((size_t)MROWS * D_MODEL * 2);
    unsigned short* vtb  = (unsigned short*)alloc((size_t)MROWS * D_MODEL * 2);
    unsigned short* ctx  = (unsigned short*)alloc((size_t)MROWS * D_MODEL * 2);
    float*          src2 = (float*)        alloc((size_t)MROWS * D_MODEL * 4);
    unsigned short* y2   = (unsigned short*)alloc((size_t)MROWS * D_MODEL * 2);
    unsigned short* hff  = (unsigned short*)alloc((size_t)MROWS * D_FF * 2);

    dim3 tb16(16, 16);
    transpose_bf16<<<dim3(D_MODEL / 16, D_MODEL / 16), tb16, 0, stream>>>(Wq, WqT, D_MODEL, D_MODEL);
    transpose_bf16<<<dim3(D_MODEL / 16, D_MODEL / 16), tb16, 0, stream>>>(Wk, WkT, D_MODEL, D_MODEL);
    transpose_bf16<<<dim3(D_MODEL / 16, D_MODEL / 16), tb16, 0, stream>>>(Wv, WvT, D_MODEL, D_MODEL);
    transpose_bf16<<<dim3(D_MODEL / 16, D_MODEL / 16), tb16, 0, stream>>>(Wo, WoT, D_MODEL, D_MODEL);
    transpose_bf16<<<dim3(D_MODEL / 16, D_FF / 16),    tb16, 0, stream>>>(W1, W1T, D_MODEL, D_FF);
    transpose_bf16<<<dim3(D_FF / 16, D_MODEL / 16),    tb16, 0, stream>>>(W2, W2T, D_FF, D_MODEL);

    ln_kernel<<<MROWS, 256, 0, stream>>>(src, g1, beta1, xh);

    gemm_qkv<<<dim3(D_MODEL / 64, MROWS / 128, 3), 128, 0, stream>>>(
        xh, WqT, WkT, WvT, bq, bk, bv, qb, kb, vtb);

    attn_kernel<<<8 * N_HEAD * 64, 32, 0, stream>>>(qb, kb, vtb, ctx);

    gemm_bias_res_f32<<<dim3(D_MODEL / 64, MROWS / 128), 128, 0, stream>>>(
        ctx, WoT, bo, src, src2, D_MODEL, D_MODEL);

    ln_kernel<<<MROWS, 256, 0, stream>>>(src2, g2, beta2, y2);

    gemm_bias_relu_bf16<<<dim3(D_FF / 64, MROWS / 128), 128, 0, stream>>>(
        y2, W1T, b1, hff, D_MODEL, D_FF);

    gemm_bias_res_f32<<<dim3(D_MODEL / 64, MROWS / 128), 128, 0, stream>>>(
        hff, W2T, b2, src2, out, D_FF, D_MODEL);
}
